// PointCloudTokenizerV2_28346784153895
// MI455X (gfx1250) — matrix operand order repaired and checked by reading, not compiled, not run on hardware
//
#include <hip/hip_runtime.h>
#include <hip/hip_bf16.h>

typedef __attribute__((ext_vector_type(16))) __bf16 v16bf;
typedef __attribute__((ext_vector_type(8)))  __bf16 v8bf;
typedef __attribute__((ext_vector_type(8)))  float  v8f;

#define N_PTS 131072
#define B_EV  32
#define P_PTS 4096
#define K_TOK 128
#define D_DIM 768

// ---------------------------------------------------------------------------
// Weight conversion: f32 [In][Out] row-major  ->  bf16 [Out][In] (transposed)
// ---------------------------------------------------------------------------
__global__ void wconv_kernel(const float* __restrict__ src, __bf16* __restrict__ dst,
                             int In, int Out) {
    int i = blockIdx.x * blockDim.x + threadIdx.x;
    int total = In * Out;
    if (i >= total) return;
    int n = i / In;
    int k = i - n * In;
    dst[i] = (__bf16)src[k * Out + n];
}

// ---------------------------------------------------------------------------
// Frontend: layernorm(4) + SE MLP (4->64 relu ->64), emit bf16 [N,128] rows
// ---------------------------------------------------------------------------
__global__ void frontend_kernel(const float* __restrict__ coords,
                                const float* __restrict__ features,
                                const float* __restrict__ times,
                                const float* __restrict__ ln_g,
                                const float* __restrict__ ln_b,
                                const float* __restrict__ se_w1,
                                const float* __restrict__ se_b1,
                                const float* __restrict__ se_w2,
                                const float* __restrict__ se_b2,
                                __bf16* __restrict__ xin) {
    __shared__ float sW1[4 * 64];
    __shared__ float sB1[64];
    __shared__ float sW2[64 * 64];
    __shared__ float sB2[64];
    __shared__ float sG[4], sBeta[4];
    int tid = threadIdx.x;
    if (tid < 256) sW1[tid] = se_w1[tid];
    if (tid < 64) { sB1[tid] = se_b1[tid]; sB2[tid] = se_b2[tid]; }
    if (tid < 4)  { sG[tid] = ln_g[tid];   sBeta[tid] = ln_b[tid]; }
    for (int i = tid; i < 64 * 64; i += 256) sW2[i] = se_w2[i];
    __syncthreads();

    int n = blockIdx.x * 256 + tid;
    float v[4];
    v[0] = coords[(size_t)n * 3 + 0];
    v[1] = coords[(size_t)n * 3 + 1];
    v[2] = coords[(size_t)n * 3 + 2];
    v[3] = times[n];
    float mu = 0.25f * (v[0] + v[1] + v[2] + v[3]);
    float var = 0.f;
    #pragma unroll
    for (int j = 0; j < 4; ++j) { float d = v[j] - mu; var += d * d; }
    var *= 0.25f;
    float inv = rsqrtf(var + 1e-5f);
    float cn[4];
    #pragma unroll
    for (int j = 0; j < 4; ++j) cn[j] = (v[j] - mu) * inv * sG[j] + sBeta[j];

    float h[64];
    #pragma unroll 8
    for (int o = 0; o < 64; ++o) {
        float a = sB1[o];
        #pragma unroll
        for (int j = 0; j < 4; ++j) a += cn[j] * sW1[j * 64 + o];
        h[o] = fmaxf(a, 0.f);
    }

    __bf16* xr = xin + (size_t)n * 128;
    for (int f = 0; f < 64; ++f) xr[f] = (__bf16)features[(size_t)n * 64 + f];
    #pragma unroll 4
    for (int o = 0; o < 64; ++o) {
        float a = sB2[o];
        for (int j = 0; j < 64; ++j) a += h[j] * sW2[j * 64 + o];
        xr[64 + o] = (__bf16)a;
    }
}

// ---------------------------------------------------------------------------
// One dense layer on a 64-row tile with v_wmma_f32_16x16x32_bf16.
// Each wave iteration computes a 16x32 output block (two accumulators) so
// independent wmma fill each other's hazard/wait slots and A LDS reads are
// amortized over two matrix ops.
// actIn : LDS bf16 [64][KIN]     actOut : LDS bf16 [64][NOUT]
// wT    : global bf16 [NOUT][KIN] (transposed weights, L2-resident)
// ---------------------------------------------------------------------------
template <int KIN, int NOUT, bool RELU>
__device__ __forceinline__ void layer_wmma(const __bf16* actIn, __bf16* actOut,
                                           const __bf16* __restrict__ wT,
                                           const float* __restrict__ bias,
                                           int wave, int lane) {
    const int lm = lane & 15;   // row (A) / col (B) within tile
    const int lh = lane >> 4;   // half-select per ISA VGPR layout
    const int nTiles = 4 * (NOUT / 32);          // strips x 32-col blocks
    for (int t = wave; t < nTiles; t += 8) {     // strip = t&3 constant per wave
        const int strip = t & 3;
        const int blk   = t >> 2;
        const int m0 = strip * 16;
        const int n0 = blk * 32;
        v8f acc0 = {};
        v8f acc1 = {};
        const __bf16* aBase  = actIn + (m0 + lm) * KIN;
        const __bf16* bBase0 = wT + (size_t)(n0 + lm) * KIN;
        const __bf16* bBase1 = wT + (size_t)(n0 + 16 + lm) * KIN;
        #pragma unroll 4
        for (int kc = 0; kc < KIN / 32; ++kc) {
            const int k0 = kc * 32;
            // A 16x32 bf16: lanes 0-15 hold K 0..15, lanes 16-31 hold K 16..31
            v8bf a0 = *(const v8bf*)(aBase + k0 + lh * 16);
            v8bf a1 = *(const v8bf*)(aBase + k0 + lh * 16 + 8);
            // B 32x16 bf16: lanes 0-15 hold K 0..15, lanes 16-31 hold K 16..31
            v8bf b00 = *(const v8bf*)(bBase0 + k0 + lh * 16);
            v8bf b01 = *(const v8bf*)(bBase0 + k0 + lh * 16 + 8);
            v8bf b10 = *(const v8bf*)(bBase1 + k0 + lh * 16);
            v8bf b11 = *(const v8bf*)(bBase1 + k0 + lh * 16 + 8);
            v16bf A  = __builtin_shufflevector(a0,  a1,  0,1,2,3,4,5,6,7,8,9,10,11,12,13,14,15);
            v16bf B0 = __builtin_shufflevector(b00, b01, 0,1,2,3,4,5,6,7,8,9,10,11,12,13,14,15);
            v16bf B1 = __builtin_shufflevector(b10, b11, 0,1,2,3,4,5,6,7,8,9,10,11,12,13,14,15);
            acc0 = __builtin_amdgcn_wmma_f32_16x16x32_bf16(
                false, A, false, B0, (short)0, acc0, false, false);
            acc1 = __builtin_amdgcn_wmma_f32_16x16x32_bf16(
                false, A, false, B1, (short)0, acc1, false, false);
        }
        const float bn0 = bias[n0 + lm];
        const float bn1 = bias[n0 + 16 + lm];
        #pragma unroll
        for (int i = 0; i < 8; ++i) {
            float v0 = acc0[i] + bn0;
            float v1 = acc1[i] + bn1;
            if (RELU) { v0 = fmaxf(v0, 0.f); v1 = fmaxf(v1, 0.f); }
            const int row = m0 + lh * 8 + i;   // C/D layout: VGPR i -> M=i (+8 for hi half)
            actOut[row * NOUT + (n0 + lm)]      = (__bf16)v0;
            actOut[row * NOUT + (n0 + 16 + lm)] = (__bf16)v1;
        }
    }
}

// ---------------------------------------------------------------------------
// Main MLP chain: 128->256->512->768->768 (+ importance 768->384->1)
// One workgroup (8 wave32) per 64-point tile; 192 KB dynamic LDS ping-pong.
// ---------------------------------------------------------------------------
__global__ void mlp_kernel(const __bf16* __restrict__ xin,
                           const __bf16* __restrict__ wT0, const __bf16* __restrict__ wT1,
                           const __bf16* __restrict__ wT2, const __bf16* __restrict__ wT3,
                           const __bf16* __restrict__ wTi,
                           const float* __restrict__ b0, const float* __restrict__ b1,
                           const float* __restrict__ b2, const float* __restrict__ b3,
                           const float* __restrict__ ib0, const float* __restrict__ iw1,
                           const float* __restrict__ ib1,
                           __bf16* __restrict__ pf, float* __restrict__ sc) {
    extern __shared__ __bf16 smem[];
    __bf16* bufA = smem;              // 64*768 bf16 = 96 KB
    __bf16* bufB = smem + 64 * 768;   // 64*768 bf16 = 96 KB
    const int tid  = threadIdx.x;
    const int wave = tid >> 5;
    const int lane = tid & 31;
    const size_t base = (size_t)blockIdx.x * 64;

    // stage 64x128 bf16 activation tile into LDS (vectorized b128)
    {
        const v8bf* src = (const v8bf*)(xin + base * 128);
        v8bf* dst = (v8bf*)bufA;
        for (int i = tid; i < 64 * 128 / 8; i += 256) dst[i] = src[i];
    }
    __syncthreads();
    layer_wmma<128, 256, true >(bufA, bufB, wT0, b0, wave, lane);
    __syncthreads();
    layer_wmma<256, 512, true >(bufB, bufA, wT1, b1, wave, lane);
    __syncthreads();
    layer_wmma<512, 768, true >(bufA, bufB, wT2, b2, wave, lane);
    __syncthreads();
    layer_wmma<768, 768, false>(bufB, bufA, wT3, b3, wave, lane);   // pf tile
    __syncthreads();

    // spill pf tile to global (bf16)
    {
        const v8bf* src = (const v8bf*)bufA;
        v8bf* dst = (v8bf*)(pf + base * 768);
        for (int i = tid; i < 64 * 768 / 8; i += 256) dst[i] = src[i];
    }

    // importance head: h = relu(pf @ imp_w0) then sc = h @ imp_w1 + b
    layer_wmma<768, 384, true>(bufA, bufB, wTi, ib0, wave, lane);
    __syncthreads();
    if (tid < 64) {
        float a = ib1[0];
        const __bf16* hr = bufB + tid * 384;
        for (int j = 0; j < 384; ++j) a += (float)hr[j] * iw1[j];
        sc[base + tid] = a;
    }
}

// ---------------------------------------------------------------------------
// Per-batch top-K (bitonic sort 4096), re-sort selected 128 by time, gather.
// ---------------------------------------------------------------------------
__global__ void topk_kernel(const float* __restrict__ sc,
                            const __bf16* __restrict__ pf,
                            const float* __restrict__ coords,
                            const float* __restrict__ times,
                            float* __restrict__ tokens,   // [B,K,768]
                            float* __restrict__ cents,    // [B,K,4]
                            float* __restrict__ mask) {   // [B,K]
    __shared__ float s[P_PTS];
    __shared__ int   ix[P_PTS];
    __shared__ float tkey[K_TOK];
    __shared__ int   tix[K_TOK];
    const int b = blockIdx.x;
    const int tid = threadIdx.x;

    for (int i = tid; i < P_PTS; i += 256) { s[i] = sc[(size_t)b * P_PTS + i]; ix[i] = i; }
    __syncthreads();

    // bitonic sort, descending by score
    for (int size = 2; size <= P_PTS; size <<= 1) {
        for (int stride = size >> 1; stride > 0; stride >>= 1) {
            for (int i = tid; i < P_PTS / 2; i += 256) {
                int pos = 2 * i - (i & (stride - 1));
                int q = pos + stride;
                bool desc = ((pos & size) == 0);
                float sa = s[pos], sb = s[q];
                bool doSwap = desc ? (sa < sb) : (sa > sb);
                if (doSwap) {
                    s[pos] = sb; s[q] = sa;
                    int ta = ix[pos]; ix[pos] = ix[q]; ix[q] = ta;
                }
            }
            __syncthreads();
        }
    }

    // re-sort top-128 ascending by time
    if (tid < K_TOK) {
        int p = ix[tid];
        tix[tid] = p;
        tkey[tid] = times[(size_t)b * P_PTS + p];
    }
    __syncthreads();
    for (int size = 2; size <= K_TOK; size <<= 1) {
        for (int stride = size >> 1; stride > 0; stride >>= 1) {
            if (tid < K_TOK / 2) {
                int pos = 2 * tid - (tid & (stride - 1));
                int q = pos + stride;
                bool asc = ((pos & size) == 0);
                float ka = tkey[pos], kb = tkey[q];
                bool doSwap = asc ? (ka > kb) : (ka < kb);
                if (doSwap) {
                    tkey[pos] = kb; tkey[q] = ka;
                    int tt = tix[pos]; tix[pos] = tix[q]; tix[q] = tt;
                }
            }
            __syncthreads();
        }
    }

    // gather outputs
    for (int e = tid; e < K_TOK * D_DIM; e += 256) {
        int k = e / D_DIM, d = e - k * D_DIM;
        int p = tix[k];
        tokens[((size_t)b * K_TOK + k) * D_DIM + d] =
            (float)pf[((size_t)b * P_PTS + p) * D_DIM + d];
    }
    for (int e = tid; e < K_TOK * 4; e += 256) {
        int k = e >> 2, c = e & 3;
        int p = tix[k];
        size_t n = (size_t)b * P_PTS + p;
        cents[((size_t)b * K_TOK + k) * 4 + c] = (c < 3) ? coords[n * 3 + c] : times[n];
    }
    for (int e = tid; e < K_TOK; e += 256) mask[(size_t)b * K_TOK + e] = 1.0f;
}

// ---------------------------------------------------------------------------
extern "C" void kernel_launch(void* const* d_in, const int* in_sizes, int n_in,
                              void* d_out, int out_size, void* d_ws, size_t ws_size,
                              hipStream_t stream) {
    const float* coords   = (const float*)d_in[0];
    const float* features = (const float*)d_in[1];
    const float* times    = (const float*)d_in[2];
    // d_in[3] batch_ids: contiguous equal groups -> implicit; d_in[4] tau unused (forward path)
    const float* ln_g  = (const float*)d_in[5];
    const float* ln_b  = (const float*)d_in[6];
    const float* se_w1 = (const float*)d_in[7];
    const float* se_b1 = (const float*)d_in[8];
    const float* se_w2 = (const float*)d_in[9];
    const float* se_b2 = (const float*)d_in[10];
    const float* mw0 = (const float*)d_in[11]; const float* mb0 = (const float*)d_in[12];
    const float* mw1 = (const float*)d_in[13]; const float* mb1 = (const float*)d_in[14];
    const float* mw2 = (const float*)d_in[15]; const float* mb2 = (const float*)d_in[16];
    const float* mw3 = (const float*)d_in[17]; const float* mb3 = (const float*)d_in[18];
    const float* iw0 = (const float*)d_in[19]; const float* ib0 = (const float*)d_in[20];
    const float* iw1 = (const float*)d_in[21]; const float* ib1 = (const float*)d_in[22];

    // workspace layout (bytes)
    char* ws = (char*)d_ws;
    __bf16* wT0 = (__bf16*)(ws + 0);                         // 256x128  ->  65536 B
    __bf16* wT1 = (__bf16*)(ws + 65536);                     // 512x256  -> 262144 B
    __bf16* wT2 = (__bf16*)(ws + 327680);                    // 768x512  -> 786432 B
    __bf16* wT3 = (__bf16*)(ws + 1114112);                   // 768x768  -> 1179648 B
    __bf16* wTi = (__bf16*)(ws + 2293760);                   // 384x768  -> 589824 B
    __bf16* xin = (__bf16*)(ws + 2883584);                   // N x128 bf16
    __bf16* pf  = (__bf16*)(ws + 2883584 + (size_t)N_PTS * 128 * 2);       // N x768 bf16
    float*  sc  = (float*)(ws + 2883584 + (size_t)N_PTS * 128 * 2
                                        + (size_t)N_PTS * 768 * 2);        // N f32

    // 1) weight transpose + bf16 conversion
    wconv_kernel<<<(128 * 256 + 255) / 256, 256, 0, stream>>>(mw0, wT0, 128, 256);
    wconv_kernel<<<(256 * 512 + 255) / 256, 256, 0, stream>>>(mw1, wT1, 256, 512);
    wconv_kernel<<<(512 * 768 + 255) / 256, 256, 0, stream>>>(mw2, wT2, 512, 768);
    wconv_kernel<<<(768 * 768 + 255) / 256, 256, 0, stream>>>(mw3, wT3, 768, 768);
    wconv_kernel<<<(768 * 384 + 255) / 256, 256, 0, stream>>>(iw0, wTi, 768, 384);

    // 2) layernorm + SE frontend
    frontend_kernel<<<N_PTS / 256, 256, 0, stream>>>(
        coords, features, times, ln_g, ln_b, se_w1, se_b1, se_w2, se_b2, xin);

    // 3) WMMA MLP chain (64-point tiles, 192 KB dynamic LDS)
    mlp_kernel<<<N_PTS / 64, 256, 2 * 64 * 768 * (int)sizeof(__bf16), stream>>>(
        xin, wT0, wT1, wT2, wT3, wTi,
        mb0, mb1, mb2, mb3, ib0, iw1, ib1, pf, sc);

    // 4) top-K + time sort + gather
    float* tokens = (float*)d_out;
    float* cents  = tokens + (size_t)B_EV * K_TOK * D_DIM;
    float* mask   = cents + (size_t)B_EV * K_TOK * 4;
    topk_kernel<<<B_EV, 256, 0, stream>>>(sc, pf, coords, times, tokens, cents, mask);
}
